// GRUTransformer_40707700032439
// MI455X (gfx1250) — compile-verified, matrix operand order checked by reading, not run on hardware
//
#include <hip/hip_runtime.h>
#include <hip/hip_bf16.h>

// ---------------- model constants ----------------
#define Bsz   256
#define Ssz   512
#define EMB   128
#define NF    64
#define INF_  320      // IN = 2*EMB + NF
#define Hd    512
#define H3    1536
#define FFN_  2048

typedef unsigned short u16;
typedef __attribute__((ext_vector_type(16))) __bf16 bf16x16;
typedef __attribute__((ext_vector_type(8)))  float  f32x8;

static __device__ __forceinline__ u16 f32_to_bf16(float f) {
    unsigned int u = __float_as_uint(f);
    u += 0x7FFFu + ((u >> 16) & 1u);   // round to nearest even
    return (u16)(u >> 16);
}

union Frag { uint4 u[2]; bf16x16 v; };

// ---------------- CDNA5 async global->LDS (ASYNCcnt-tracked) ----------------
static __device__ __forceinline__ void async_b128(unsigned ldsAddr, const u16* g) {
    asm volatile("global_load_async_to_lds_b128 %0, %1, off"
                 :: "v"(ldsAddr), "v"((unsigned long long)(__SIZE_TYPE__)g)
                 : "memory");
}
template<int N>
static __device__ __forceinline__ void wait_asynccnt() {
    asm volatile("s_wait_asynccnt %0" :: "i"(N) : "memory");
}

// ---------------- f32 -> bf16 convert ----------------
__global__ __launch_bounds__(256) void conv_bf16_kernel(const float* __restrict__ src,
                                                        u16* __restrict__ dst, long n) {
    long i = (long)blockIdx.x * blockDim.x + threadIdx.x;
    if (i < n) dst[i] = f32_to_bf16(src[i]);
}

// ---------------- pack GRU weight (3H x K) into per-j-tile slabs [jt][r|z|n][64][K], bf16 ----------------
__global__ __launch_bounds__(256) void pack_w_kernel(const float* __restrict__ W,
                                                     u16* __restrict__ Wp, int K) {
    long i = (long)blockIdx.x * 256 + threadIdx.x;   // over 1536*K
    int  k = (int)(i % K);
    long rp = i / K;
    int jt = (int)(rp / 192), rem = (int)(rp % 192);
    int c = rem / 64, r = rem % 64;
    long src = ((long)(c * Hd + jt * 64 + r)) * K + k;
    Wp[i] = f32_to_bf16(W[src]);
}

// ---------------- embedding + RBF features (bf16) ----------------
__global__ __launch_bounds__(256) void feats_kernel(const int* __restrict__ a1,
                                                    const int* __restrict__ a2,
                                                    const float* __restrict__ bond,
                                                    const float* __restrict__ emb,
                                                    u16* __restrict__ feats) {
    long i = (long)blockIdx.x * blockDim.x + threadIdx.x;      // over B*S*IN
    long bs = i / INF_;
    int  e  = (int)(i - bs * INF_);
    float v;
    if (e < EMB)            v = emb[(long)a1[bs] * EMB + e];
    else if (e < 2 * EMB)   v = emb[(long)a2[bs] * EMB + (e - EMB)];
    else {
        float c = (float)(e - 2 * EMB + 1);                    // centers 1..64
        float d = c - bond[bs];
        v = __expf(-d * d);                                    // GAMMA = 1
    }
    feats[i] = f32_to_bf16(v);
}

// ================= fused GRU step: h_out = GRUgate(x@Wi^T+bi, h@Wh^T+bh, h_in) =================
// Block computes 32 batch rows x 64 j-cols of h. Weights pre-permuted so the r/z/n rows of a
// j-tile are one contiguous 192xK slab: block runs 2 async-pipelined K loops (x-part, h-part)
// holding 6 accumulator tiles, then applies the gate in-register. 2 graph nodes per GRU step.
#define GM 32
#define GN 64
#define GK 64
#define TROWS (GM + 3 * GN)   // 224 LDS rows: [0,32) = A, [32,224) = W slab

struct GruArgs {
    const u16* Ax; int lda; int Kx;     // x input (bf16), row stride, K
    const u16* Hb;                      // h_in bf16 (stride Hd)
    const u16* Wip;                     // packed Wi slabs [8][192][Kx]
    const u16* Whp;                     // packed Wh slabs [8][192][Hd]
    const float* bi; const float* bh;   // 3H biases (unpermuted)
    const float* hin;                   // h_in f32
    float* hout; u16* houtb;            // h_out f32 + bf16
};

__global__ __launch_bounds__(256) void gru_step_kernel(GruArgs g) {
    __shared__ u16 Ts[2][TROWS * GK];   // 2 x 28 KB
    const int tid  = threadIdx.x;
    const int wave = tid >> 5, lane = tid & 31;
    const int wm = wave >> 2, wn = wave & 3;        // 2(M) x 4(N) waves, 16x16 tiles
    const int m0 = blockIdx.y * GM;                 // batch rows
    const int jt = blockIdx.x;                      // j tile
    const int j0 = jt * GN;
    const int lr = lane & 15, lh = lane >> 4;

    // per-thread static chunk map: 224 rows * 8 chunks = 1792 = 7 * 256
    int rr[7], c8[7];
    #pragma unroll
    for (int i = 0; i < 7; ++i) { int c = tid + i * 256; rr[i] = c >> 3; c8[i] = (c & 7) * 8; }

    f32x8 accX[3] = {}, accH[3] = {};

    auto run_phase = [&](const u16* Abase, int lda, const u16* Wslab, int K, f32x8* acc) {
        const u16* gp[7];
        #pragma unroll
        for (int i = 0; i < 7; ++i)
            gp[i] = (rr[i] < GM) ? (Abase + (__SIZE_TYPE__)(m0 + rr[i]) * lda + c8[i])
                                 : (Wslab + (__SIZE_TYPE__)(rr[i] - GM) * K + c8[i]);
        auto issue = [&](int bufi) {
            #pragma unroll
            for (int i = 0; i < 7; ++i) {
                async_b128((unsigned)(__SIZE_TYPE__)&Ts[bufi][rr[i] * GK + c8[i]], gp[i]);
                gp[i] += GK;
            }
        };
        const int nIter = K / GK;
        issue(0);
        for (int it = 0; it < nIter; ++it) {
            const int buf = it & 1;
            if (it + 1 < nIter) { issue(buf ^ 1); wait_asynccnt<7>(); }
            else                { wait_asynccnt<0>(); }
            __syncthreads();
            #pragma unroll
            for (int kk = 0; kk < 2; ++kk) {
                Frag a;
                const u16* ab = &Ts[buf][(wm * 16 + lr) * GK + kk * 32];
                a.u[0] = *(const uint4*)(ab + lh * 8);
                a.u[1] = *(const uint4*)(ab + 16 + lh * 8);
                #pragma unroll
                for (int c = 0; c < 3; ++c) {
                    Frag w;
                    const u16* wb = &Ts[buf][(GM + c * GN + wn * 16 + lr) * GK + kk * 32];
                    w.u[0] = *(const uint4*)(wb + lh * 8);
                    w.u[1] = *(const uint4*)(wb + 16 + lh * 8);
                    acc[c] = __builtin_amdgcn_wmma_f32_16x16x32_bf16(
                        false, a.v, false, w.v, (short)0, acc[c], false, false);
                }
            }
            __syncthreads();
        }
    };

    run_phase(g.Ax, g.lda, g.Wip + (__SIZE_TYPE__)jt * 192 * g.Kx, g.Kx, accX);
    run_phase(g.Hb, Hd,    g.Whp + (__SIZE_TYPE__)jt * 192 * Hd,   Hd,   accH);

    // in-register GRU gate (reference-exact math)
    const int colj = j0 + wn * 16 + lr;                 // j within H
    const float bir = g.bi[colj],          bhr = g.bh[colj];
    const float biz = g.bi[Hd + colj],     bhz = g.bh[Hd + colj];
    const float bin = g.bi[2 * Hd + colj], bhn = g.bh[2 * Hd + colj];
    #pragma unroll
    for (int i = 0; i < 8; ++i) {
        int row = m0 + wm * 16 + lh * 8 + i;
        float Xr = accX[0][i] + bir, Hr = accH[0][i] + bhr;
        float Xz = accX[1][i] + biz, Hz = accH[1][i] + bhz;
        float Xn = accX[2][i] + bin, Hn = accH[2][i] + bhn;
        float sr = 1.f / (1.f + __expf(-(Xr + Hr)));
        float sz = 1.f / (1.f + __expf(-(Xz + Hz)));
        float r = __expf(-__expf(-sr));
        float z = __expf(-__expf(-sz));
        float n = tanhf(Xn + Hn + r * Hn);
        float hold = g.hin[(long)row * Hd + colj];
        float out = (1.f - z) * n + z * hold;
        g.hout[(long)row * Hd + colj]  = out;
        g.houtb[(long)row * Hd + colj] = f32_to_bf16(out);
    }
}

// ================= generic WMMA GEMM (transformer / head) =================
// C[M,N] = A[M,K] * W[N,K]^T (+bias, ReLU). Block 128x64, BK=64, double-buffered async LDS.
#define BM 128
#define BN 64
#define BK 64

struct GemmArgs {
    const u16* A; int lda;
    const u16* W;
    const float* bias;
    float* C; u16* Cbf;
    int N; int K; int relu;
};

__global__ __launch_bounds__(256) void gemm_one(GemmArgs g) {
    __shared__ u16 As[2][BM * BK];
    __shared__ u16 Ws[2][BN * BK];
    const int tid  = threadIdx.x;
    const int wave = tid >> 5, lane = tid & 31;
    const int wm = wave >> 1, wn = wave & 1;
    const int m0 = blockIdx.y * BM;
    const int n0 = blockIdx.x * BN;
    const int lr = lane & 15, lh = lane >> 4;

    int arow[4], acol[4];
    #pragma unroll
    for (int i = 0; i < 4; ++i) { int c = tid + i * 256; arow[i] = c >> 3; acol[i] = (c & 7) * 8; }
    int wrow[2], wcol[2];
    #pragma unroll
    for (int i = 0; i < 2; ++i) { int c = tid + i * 256; wrow[i] = c >> 3; wcol[i] = (c & 7) * 8; }

    const u16* gA[4];
    #pragma unroll
    for (int i = 0; i < 4; ++i) gA[i] = g.A + (__SIZE_TYPE__)(m0 + arow[i]) * g.lda + acol[i];
    const u16* gW[2];
    #pragma unroll
    for (int i = 0; i < 2; ++i) gW[i] = g.W + (__SIZE_TYPE__)(n0 + wrow[i]) * g.K + wcol[i];

    auto issue = [&](int bufi) {
        #pragma unroll
        for (int i = 0; i < 4; ++i) {
            async_b128((unsigned)(__SIZE_TYPE__)&As[bufi][arow[i] * BK + acol[i]], gA[i]);
            gA[i] += BK;
        }
        #pragma unroll
        for (int i = 0; i < 2; ++i) {
            async_b128((unsigned)(__SIZE_TYPE__)&Ws[bufi][wrow[i] * BK + wcol[i]], gW[i]);
            gW[i] += BK;
        }
    };

    f32x8 acc[2][2] = {};
    const int nIter = g.K / BK;
    issue(0);
    for (int it = 0; it < nIter; ++it) {
        const int buf = it & 1;
        if (it + 1 < nIter) { issue(buf ^ 1); wait_asynccnt<6>(); }
        else                { wait_asynccnt<0>(); }
        __syncthreads();
        #pragma unroll
        for (int kk = 0; kk < 2; ++kk) {
            Frag a[2], b[2];
            #pragma unroll
            for (int sm = 0; sm < 2; ++sm) {
                const u16* base = &As[buf][(wm * 32 + sm * 16 + lr) * BK + kk * 32];
                a[sm].u[0] = *(const uint4*)(base + lh * 8);
                a[sm].u[1] = *(const uint4*)(base + 16 + lh * 8);
            }
            #pragma unroll
            for (int sn = 0; sn < 2; ++sn) {
                const u16* base = &Ws[buf][(wn * 32 + sn * 16 + lr) * BK + kk * 32];
                b[sn].u[0] = *(const uint4*)(base + lh * 8);
                b[sn].u[1] = *(const uint4*)(base + 16 + lh * 8);
            }
            #pragma unroll
            for (int sm = 0; sm < 2; ++sm)
                #pragma unroll
                for (int sn = 0; sn < 2; ++sn)
                    acc[sm][sn] = __builtin_amdgcn_wmma_f32_16x16x32_bf16(
                        false, a[sm].v, false, b[sn].v, (short)0, acc[sm][sn], false, false);
        }
        __syncthreads();
    }

    const int mb = lh * 8;
    #pragma unroll
    for (int sm = 0; sm < 2; ++sm)
        #pragma unroll
        for (int sn = 0; sn < 2; ++sn) {
            int col = n0 + wn * 32 + sn * 16 + lr;
            float bv = g.bias ? g.bias[col] : 0.f;
            #pragma unroll
            for (int i = 0; i < 8; ++i) {
                int row = m0 + wm * 32 + sm * 16 + mb + i;
                float v = acc[sm][sn][i] + bv;
                if (g.relu) v = fmaxf(v, 0.f);
                g.C[(long)row * g.N + col] = v;
                if (g.Cbf) g.Cbf[(long)row * g.N + col] = f32_to_bf16(v);
            }
        }
}

// ---------------- residual add + LayerNorm (one block per row) ----------------
__global__ __launch_bounds__(256) void add_ln_kernel(float* __restrict__ x,
                                                     u16* __restrict__ xbf,
                                                     const float* __restrict__ o,
                                                     const float* __restrict__ g,
                                                     const float* __restrict__ bt) {
    __shared__ float red[256];
    int b = blockIdx.x, tid = threadIdx.x;
    long base = (long)b * Hd;
    float v0 = x[base + tid]       + o[base + tid];
    float v1 = x[base + 256 + tid] + o[base + 256 + tid];
    red[tid] = v0 + v1;
    __syncthreads();
    for (int s = 128; s > 0; s >>= 1) { if (tid < s) red[tid] += red[tid + s]; __syncthreads(); }
    float mean = red[0] * (1.f / Hd);
    __syncthreads();
    float d0 = v0 - mean, d1 = v1 - mean;
    red[tid] = d0 * d0 + d1 * d1;
    __syncthreads();
    for (int s = 128; s > 0; s >>= 1) { if (tid < s) red[tid] += red[tid + s]; __syncthreads(); }
    float rstd = rsqrtf(red[0] * (1.f / Hd) + 1e-5f);
    float y0 = d0 * rstd * g[tid] + bt[tid];
    float y1 = d1 * rstd * g[256 + tid] + bt[256 + tid];
    x[base + tid] = y0;           x[base + 256 + tid] = y1;
    xbf[base + tid] = f32_to_bf16(y0); xbf[base + 256 + tid] = f32_to_bf16(y1);
}

// ---------------- final head dot ----------------
__global__ __launch_bounds__(256) void final_dot_kernel(const float* __restrict__ G,
                                                        const float* __restrict__ w2,
                                                        const float* __restrict__ b2,
                                                        float* __restrict__ out) {
    __shared__ float red[256];
    int b = blockIdx.x, tid = threadIdx.x;
    float acc = 0.f;
    for (int j = tid; j < FFN_; j += 256) acc += G[(long)b * FFN_ + j] * w2[j];
    red[tid] = acc;
    __syncthreads();
    for (int s = 128; s > 0; s >>= 1) { if (tid < s) red[tid] += red[tid + s]; __syncthreads(); }
    if (tid == 0) out[b] = red[0] + b2[0];
}

// ---------------- launch ----------------
extern "C" void kernel_launch(void* const* d_in, const int* in_sizes, int n_in,
                              void* d_out, int out_size, void* d_ws, size_t ws_size,
                              hipStream_t stream) {
    (void)in_sizes; (void)n_in; (void)out_size; (void)ws_size;
    const int*   atom1 = (const int*)  d_in[0];
    const int*   atom2 = (const int*)  d_in[1];
    const float* bond  = (const float*)d_in[2];
    const float* emb   = (const float*)d_in[3];
    const float* gWi0  = (const float*)d_in[4];
    const float* gbi0  = (const float*)d_in[5];
    const float* gWh0  = (const float*)d_in[6];
    const float* gbh0  = (const float*)d_in[7];
    const float* gWi1  = (const float*)d_in[8];
    const float* gbi1  = (const float*)d_in[9];
    const float* gWh1  = (const float*)d_in[10];
    const float* gbh1  = (const float*)d_in[11];
    const float* qkv_w = (const float*)d_in[12];
    const float* qkv_b = (const float*)d_in[13];
    const float* out_w = (const float*)d_in[14];
    const float* out_b = (const float*)d_in[15];
    const float* ln1_g = (const float*)d_in[16];
    const float* ln1_b = (const float*)d_in[17];
    const float* ln2_g = (const float*)d_in[18];
    const float* ln2_b = (const float*)d_in[19];
    const float* ffn_w1 = (const float*)d_in[20];
    const float* ffn_b1 = (const float*)d_in[21];
    const float* ffn_w2 = (const float*)d_in[22];
    const float* ffn_b2 = (const float*)d_in[23];
    const float* fc_w1 = (const float*)d_in[24];
    const float* fc_b1 = (const float*)d_in[25];
    const float* fc_w2 = (const float*)d_in[26];
    const float* fc_b2 = (const float*)d_in[27];
    float* out = (float*)d_out;

    // ---- workspace layout ----
    size_t off = 0;
    auto ALLOC = [&](size_t bytes) -> void* {
        void* p = (char*)d_ws + off;
        off = (off + bytes + 255) & ~(size_t)255;
        return p;
    };
    u16* wWi0p = (u16*)ALLOC((size_t)H3 * INF_ * 2);   // packed slabs
    u16* wWh0p = (u16*)ALLOC((size_t)H3 * Hd * 2);
    u16* wWi1p = (u16*)ALLOC((size_t)H3 * Hd * 2);
    u16* wWh1p = (u16*)ALLOC((size_t)H3 * Hd * 2);
    u16* wQKV  = (u16*)ALLOC((size_t)2 * H3 * Hd * 2);
    u16* wOut  = (u16*)ALLOC((size_t)2 * Hd * Hd * 2);
    u16* wF1   = (u16*)ALLOC((size_t)2 * FFN_ * Hd * 2);
    u16* wF2   = (u16*)ALLOC((size_t)2 * Hd * FFN_ * 2);
    u16* wFC1  = (u16*)ALLOC((size_t)FFN_ * Hd * 2);
    u16* feats = (u16*)ALLOC((size_t)Bsz * Ssz * INF_ * 2);
    float* h0f[2]; u16* h0b_[2]; float* h1f[2]; u16* h1b_[2];
    for (int p = 0; p < 2; ++p) {
        h0f[p]  = (float*)ALLOC((size_t)Bsz * Hd * 4);
        h0b_[p] = (u16*)  ALLOC((size_t)Bsz * Hd * 2);
        h1f[p]  = (float*)ALLOC((size_t)Bsz * Hd * 4);
        h1b_[p] = (u16*)  ALLOC((size_t)Bsz * Hd * 2);
    }
    float* tV  = (float*)ALLOC((size_t)Bsz * Hd * 4); u16* tVb = (u16*)ALLOC((size_t)Bsz * Hd * 2);
    float* Op  = (float*)ALLOC((size_t)Bsz * Hd * 4);
    float* F1  = (float*)ALLOC((size_t)Bsz * FFN_ * 4); u16* F1b = (u16*)ALLOC((size_t)Bsz * FFN_ * 2);
    float* F2  = (float*)ALLOC((size_t)Bsz * Hd * 4);

    // ---- weight prep ----
    auto PACK = [&](const float* s, u16* d, int K) {
        pack_w_kernel<<<(unsigned)(((long)H3 * K) / 256), 256, 0, stream>>>(s, d, K);
    };
    PACK(gWi0, wWi0p, INF_);
    PACK(gWh0, wWh0p, Hd);
    PACK(gWi1, wWi1p, Hd);
    PACK(gWh1, wWh1p, Hd);
    auto CONV = [&](const float* s, u16* d, long n) {
        conv_bf16_kernel<<<(unsigned)((n + 255) / 256), 256, 0, stream>>>(s, d, n);
    };
    CONV(qkv_w, wQKV, (long)2 * H3 * Hd);
    CONV(out_w, wOut, (long)2 * Hd * Hd);
    CONV(ffn_w1, wF1, (long)2 * FFN_ * Hd);
    CONV(ffn_w2, wF2, (long)2 * Hd * FFN_);
    CONV(fc_w1, wFC1, (long)FFN_ * Hd);

    // ---- features ----
    {
        long n = (long)Bsz * Ssz * INF_;
        feats_kernel<<<(unsigned)(n / 256), 256, 0, stream>>>(atom1, atom2, bond, emb, feats);
    }

    // ---- init hidden state (t=0 read buffers = index 0) ----
    hipMemsetAsync(h0f[0], 0, (size_t)Bsz * Hd * 4, stream);
    hipMemsetAsync(h0b_[0], 0, (size_t)Bsz * Hd * 2, stream);
    hipMemsetAsync(h1f[0], 0, (size_t)Bsz * Hd * 4, stream);
    hipMemsetAsync(h1b_[0], 0, (size_t)Bsz * Hd * 2, stream);

    // ---- GRU scan: 2 fused nodes per step ----
    for (int t = 0; t < Ssz; ++t) {
        const int p = t & 1, q = p ^ 1;
        GruArgs a0{feats + (size_t)t * INF_, Ssz * INF_, INF_,
                   h0b_[p], wWi0p, wWh0p, gbi0, gbh0, h0f[p], h0f[q], h0b_[q]};
        gru_step_kernel<<<dim3(Hd / GN, Bsz / GM), 256, 0, stream>>>(a0);
        GruArgs a1{h0b_[q], Hd, Hd,
                   h1b_[p], wWi1p, wWh1p, gbi1, gbh1, h1f[p], h1f[q], h1b_[q]};
        gru_step_kernel<<<dim3(Hd / GN, Bsz / GM), 256, 0, stream>>>(a1);
    }
    // after t=511 (odd), new state is in buffer index 0
    float* x  = h1f[0];
    u16*   xb = h1b_[0];

    auto GEMM1 = [&](const u16* A, int lda, const u16* W, const float* bias,
                     float* C, u16* Cbf, int N, int K, int relu) {
        GemmArgs g{A, lda, W, bias, C, Cbf, N, K, relu};
        gemm_one<<<dim3((unsigned)(N / BN), Bsz / BM, 1), 256, 0, stream>>>(g);
    };

    // ---- transformer (seq len 1: softmax over one key == 1, so o = V) ----
    for (int l = 0; l < 2; ++l) {
        const u16*   wV  = wQKV + ((size_t)l * H3 + 2 * Hd) * Hd;   // V rows of qkv_w[l]
        const float* bV  = qkv_b + (size_t)l * H3 + 2 * Hd;
        GEMM1(xb, Hd, wV, bV, tV, tVb, Hd, Hd, 0);
        GEMM1(tVb, Hd, wOut + (size_t)l * Hd * Hd, out_b + (size_t)l * Hd, Op, nullptr, Hd, Hd, 0);
        add_ln_kernel<<<Bsz, 256, 0, stream>>>(x, xb, Op, ln1_g + (size_t)l * Hd, ln1_b + (size_t)l * Hd);
        GEMM1(xb, Hd, wF1 + (size_t)l * FFN_ * Hd, ffn_b1 + (size_t)l * FFN_, F1, F1b, FFN_, Hd, 1);
        GEMM1(F1b, FFN_, wF2 + (size_t)l * Hd * FFN_, ffn_b2 + (size_t)l * Hd, F2, nullptr, Hd, FFN_, 0);
        add_ln_kernel<<<Bsz, 256, 0, stream>>>(x, xb, F2, ln2_g + (size_t)l * Hd, ln2_b + (size_t)l * Hd);
    }

    // ---- head ----
    GEMM1(xb, Hd, wFC1, fc_b1, F1, nullptr, FFN_, Hd, 1);
    final_dot_kernel<<<Bsz, 256, 0, stream>>>(F1, fc_w2, fc_b2, out);
}
